// ScannedMultiLayerLSTM_65438121722411
// MI455X (gfx1250) — matrix operand order repaired and checked by reading, not compiled, not run on hardware
//
#include <hip/hip_runtime.h>

// ---------------------------------------------------------------------------
// ScannedMultiLayerLSTM on MI455X (gfx1250, wave32, WMMA)
// T=256, B=128, W=1024, L=2.  z = [x;h] @ Wcat + b   (M=128, N=4096, K=2048)
// bf16 weights/activations, f32 accumulation via v_wmma_f32_16x16x32_bf16.
// Gate-interleaved weight layout (col = 4*n + gate) -> GEMM + LSTM cell fused.
// All A-tile staging is global_load_async_to_lds_b128 (ASYNCcnt); xs is
// pre-converted to bf16 in a parallel prologue, off the sequential scan path.
// Reset masking is applied per lane to the A fragment (row m = 16*wv + nl).
// ---------------------------------------------------------------------------

#define T_STEPS 256
#define BATCH   128
#define WIDTH   1024
#define LAYERS  2
#define KDIM    (2 * WIDTH)          // 2048 (x ++ h)
#define NDIM    (4 * WIDTH)          // 4096 (4 gates, interleaved)
#define BW      (BATCH * WIDTH)      // 131072

typedef __bf16 v16bf __attribute__((ext_vector_type(16)));
typedef float  v8f   __attribute__((ext_vector_type(8)));

union Frag16 { v16bf bf; uint4 u[2]; };

__device__ __forceinline__ unsigned short f32_to_bf16(float f) {
    unsigned int u = __float_as_uint(f);
    u += 0x7FFFu + ((u >> 16) & 1u);      // round-to-nearest-even
    return (unsigned short)(u >> 16);
}
__device__ __forceinline__ unsigned pack_bf16x2(float lo, float hi) {
    return ((unsigned)f32_to_bf16(hi) << 16) | (unsigned)f32_to_bf16(lo);
}
__device__ __forceinline__ float sigmoidf_fast(float x) {
    return 1.0f / (1.0f + __expf(-x));
}
// Low 32 bits of a generic LDS pointer == LDS byte offset (ISA 10.2).
__device__ __forceinline__ unsigned lds_off32(const void* p) {
    return (unsigned)(unsigned long long)p;
}
// Async copy of 32 bytes (16 bf16) from global to LDS: 2 x b128, one per lane.
__device__ __forceinline__ void async_copy_32B(unsigned lds_byte,
                                               const unsigned short* gsrc) {
    asm volatile(
        "global_load_async_to_lds_b128 %0, %2, off\n\t"
        "global_load_async_to_lds_b128 %1, %3, off"
        :
        : "v"(lds_byte), "v"(lds_byte + 16u), "v"(gsrc), "v"(gsrc + 8)
        : "memory");
}
__device__ __forceinline__ void wait_async0() {
    asm volatile("s_wait_asynccnt 0x0" ::: "memory");
}

// ---------------------------------------------------------------------------
// Prologue 1: fuse + permute + convert weights to bf16.
// wcat[l][k][4*n + gate] = (k<W ? Wx[l][k][gate*W+n] : Wh[l][k-W][gate*W+n])
// ---------------------------------------------------------------------------
__global__ void convert_weights_kernel(const float* __restrict__ Wx,
                                       const float* __restrict__ Wh,
                                       unsigned short* __restrict__ wcat) {
    const long total = (long)LAYERS * KDIM * NDIM;
    for (long idx = (long)blockIdx.x * blockDim.x + threadIdx.x; idx < total;
         idx += (long)gridDim.x * blockDim.x) {
        const int  l   = (int)(idx / ((long)KDIM * NDIM));
        const long rem = idx % ((long)KDIM * NDIM);
        const int  k   = (int)(rem / NDIM);
        const int  col = (int)(rem % NDIM);
        const int  n    = col >> 2;
        const int  gate = col & 3;
        const int  srccol = gate * WIDTH + n;
        float v;
        if (k < WIDTH) v = Wx[((size_t)l * WIDTH + k) * NDIM + srccol];
        else           v = Wh[((size_t)l * WIDTH + (k - WIDTH)) * NDIM + srccol];
        wcat[idx] = f32_to_bf16(v);
    }
}

// ---------------------------------------------------------------------------
// Prologue 2: convert the whole xs tensor to bf16 (parallel, off the scan's
// sequential critical path; each element is consumed exactly once).
// ---------------------------------------------------------------------------
__global__ void convert_xs_kernel(const float* __restrict__ xs,
                                  unsigned* __restrict__ xbf) {   // packed bf16x2
    const long total4 = (long)T_STEPS * BW / 4;
    for (long i = (long)blockIdx.x * blockDim.x + threadIdx.x; i < total4;
         i += (long)gridDim.x * blockDim.x) {
        const float4 f = reinterpret_cast<const float4*>(xs)[i];
        xbf[2 * i]     = pack_bf16x2(f.x, f.y);
        xbf[2 * i + 1] = pack_bf16x2(f.z, f.w);
    }
}

// ---------------------------------------------------------------------------
// Prologue 3: permute bias, zero h (both parities) and c carries.
// ---------------------------------------------------------------------------
__global__ void init_states_kernel(const float* __restrict__ b,
                                   float* __restrict__ bias_perm,
                                   unsigned short* __restrict__ hbuf,   // 2*L*B*W bf16
                                   float* __restrict__ cbuf) {          // L*B*W fp32
    const int idx = blockIdx.x * blockDim.x + threadIdx.x;
    if (idx < LAYERS * NDIM) {
        const int l = idx / NDIM, col = idx % NDIM;
        bias_perm[idx] = b[l * NDIM + (col & 3) * WIDTH + (col >> 2)];
    }
    if (idx < 2 * LAYERS * BW) hbuf[idx] = 0;
    if (idx < LAYERS * BW)     cbuf[idx] = 0.0f;
}

// ---------------------------------------------------------------------------
// Fused GEMM + LSTM cell for one (t, layer).
// Grid: 64 blocks (each owns 64 N-cols == 16 hidden units), 256 threads.
// Wave w computes M rows [16w,16w+16) x 64 cols -> 4 v8f accumulators.
// ---------------------------------------------------------------------------
template <bool LAST>
__global__ __launch_bounds__(256)
void lstm_step_kernel(const unsigned short* __restrict__ x_in,    // B x W bf16
                      const unsigned char* __restrict__ resets_t, // B  (jnp bool_)
                      const unsigned short* __restrict__ wcat_l,  // K x N bf16
                      const float* __restrict__ bias_l,           // N fp32 (permuted)
                      const unsigned short* __restrict__ h_in,    // B x W bf16
                      unsigned short* __restrict__ h_out,         // B x W bf16
                      float* __restrict__ c_state_l,              // B x W fp32
                      unsigned short* __restrict__ ymid_out,      // B x W bf16 (!LAST)
                      float* __restrict__ out_t) {                // B x W fp32 (LAST)
    __shared__ alignas(16) unsigned short ldsA[BATCH][40]; // 128 x 32 bf16, padded
    __shared__ alignas(16) unsigned short ldsBt[64][40];   // B tile transposed [n][k]
    __shared__ float ldsZ[BATCH][68];                      // 128 x 64 f32, padded

    const int tid = threadIdx.x;
    const int wv  = tid >> 5;       // wave 0..7
    const int ln  = tid & 31;       // lane 0..31
    const int g   = ln >> 4;        // lane half-group
    const int nl  = ln & 15;        // row (A) / col (B,C) within tile
    const int bx  = blockIdx.x;     // N block: cols [64*bx, 64*bx+64)
    const int colbase = bx * 64;

    v8f acc0 = {}, acc1 = {}, acc2 = {}, acc3 = {};

    // A-fill indices: 16 elems (32B) per thread, async DMA to LDS
    const int a_m  = tid >> 1;
    const int a_k0 = (tid & 1) * 16;
    const unsigned a_lds = lds_off32(&ldsA[a_m][a_k0]);
    const unsigned short* a_xsrc = x_in + (size_t)a_m * WIDTH + a_k0;
    const unsigned short* a_hsrc = h_in + (size_t)a_m * WIDTH + a_k0;
    // B-fill indices: 8 elems/thread
    const int b_kk = tid >> 3;
    const int b_c0 = (tid & 7) * 8;
    // Per-lane reset mask for the A fragment rows (row m = 16*wv + nl).
    const unsigned rowmask = resets_t[16 * wv + nl] ? 0u : ~0u;

    for (int kb = 0; kb < KDIM / 32; ++kb) {
        const int k0 = kb * 32;
        const bool h_chunk = (k0 >= WIDTH);

        // ---- issue async A-tile DMA (128 x 32 bf16), overlapped with B fill
        async_copy_32B(a_lds, h_chunk ? (a_hsrc + (k0 - WIDTH)) : (a_xsrc + k0));

        // ---- stage B tile transposed (64 x 32 bf16) through VGPRs ----
        {
            const unsigned short* src =
                wcat_l + (size_t)(k0 + b_kk) * NDIM + colbase + b_c0;
#pragma unroll
            for (int q = 0; q < 8; ++q) ldsBt[b_c0 + q][b_kk] = src[q];
            if (kb + 1 < KDIM / 32)   // prefetch next weight chunk into cache
                __builtin_prefetch(wcat_l + (size_t)(k0 + 32 + b_kk) * NDIM +
                                       colbase + b_c0, 0, 1);
        }
        wait_async0();
        __syncthreads();

        // ---- WMMA: wave wv does 16x64 ----
        Frag16 fa;
        fa.u[0] = *reinterpret_cast<const uint4*>(&ldsA[16 * wv + nl][g * 16]);
        fa.u[1] = *reinterpret_cast<const uint4*>(&ldsA[16 * wv + nl][g * 16 + 8]);
        if (h_chunk) {   // zero rows whose carry is reset at this timestep
            fa.u[0].x &= rowmask; fa.u[0].y &= rowmask;
            fa.u[0].z &= rowmask; fa.u[0].w &= rowmask;
            fa.u[1].x &= rowmask; fa.u[1].y &= rowmask;
            fa.u[1].z &= rowmask; fa.u[1].w &= rowmask;
        }

        Frag16 fb0, fb1, fb2, fb3;
        fb0.u[0] = *reinterpret_cast<const uint4*>(&ldsBt[nl][g * 16]);
        fb0.u[1] = *reinterpret_cast<const uint4*>(&ldsBt[nl][g * 16 + 8]);
        fb1.u[0] = *reinterpret_cast<const uint4*>(&ldsBt[16 + nl][g * 16]);
        fb1.u[1] = *reinterpret_cast<const uint4*>(&ldsBt[16 + nl][g * 16 + 8]);
        fb2.u[0] = *reinterpret_cast<const uint4*>(&ldsBt[32 + nl][g * 16]);
        fb2.u[1] = *reinterpret_cast<const uint4*>(&ldsBt[32 + nl][g * 16 + 8]);
        fb3.u[0] = *reinterpret_cast<const uint4*>(&ldsBt[48 + nl][g * 16]);
        fb3.u[1] = *reinterpret_cast<const uint4*>(&ldsBt[48 + nl][g * 16 + 8]);

        acc0 = __builtin_amdgcn_wmma_f32_16x16x32_bf16(false, fa.bf, false, fb0.bf,
                                                       (short)0, acc0, false, false);
        acc1 = __builtin_amdgcn_wmma_f32_16x16x32_bf16(false, fa.bf, false, fb1.bf,
                                                       (short)0, acc1, false, false);
        acc2 = __builtin_amdgcn_wmma_f32_16x16x32_bf16(false, fa.bf, false, fb2.bf,
                                                       (short)0, acc2, false, false);
        acc3 = __builtin_amdgcn_wmma_f32_16x16x32_bf16(false, fa.bf, false, fb3.bf,
                                                       (short)0, acc3, false, false);
        __syncthreads();
    }

    // ---- dump accumulators to LDS (C layout: VGPR r -> M = r + 8*g, N = nl) ----
#pragma unroll
    for (int r = 0; r < 8; ++r) {
        const int m = 16 * wv + r + 8 * g;
        ldsZ[m][ 0 + nl] = acc0[r];
        ldsZ[m][16 + nl] = acc1[r];
        ldsZ[m][32 + nl] = acc2[r];
        ldsZ[m][48 + nl] = acc3[r];
    }
    __syncthreads();

    // ---- fused LSTM cell: 16 hidden units x 128 batch, 8 per thread ----
    const int nbase = bx * 16;
#pragma unroll
    for (int q = 0; q < 8; ++q) {
        const int idx = tid * 8 + q;
        const int m = idx >> 4;
        const int u = idx & 15;
        const int n = nbase + u;
        const float iv = ldsZ[m][4 * u + 0] + bias_l[colbase + 4 * u + 0];
        const float fv = ldsZ[m][4 * u + 1] + bias_l[colbase + 4 * u + 1];
        const float gv = ldsZ[m][4 * u + 2] + bias_l[colbase + 4 * u + 2];
        const float ov = ldsZ[m][4 * u + 3] + bias_l[colbase + 4 * u + 3];
        const float i_ = sigmoidf_fast(iv);
        const float f_ = sigmoidf_fast(fv);
        const float g_ = tanhf(gv);
        const float o_ = sigmoidf_fast(ov);
        const float cp = resets_t[m] ? 0.0f : c_state_l[(size_t)m * WIDTH + n];
        const float c  = f_ * cp + i_ * g_;
        const float h  = o_ * tanhf(c);
        c_state_l[(size_t)m * WIDTH + n] = c;
        h_out[(size_t)m * WIDTH + n] = f32_to_bf16(h);
        if constexpr (LAST) out_t[(size_t)m * WIDTH + n] = h;
        else                ymid_out[(size_t)m * WIDTH + n] = f32_to_bf16(h);
    }
}

// ---------------------------------------------------------------------------
// Host launcher. d_in order: xs, resets, Wx, Wh, b
// ---------------------------------------------------------------------------
extern "C" void kernel_launch(void* const* d_in, const int* in_sizes, int n_in,
                              void* d_out, int out_size, void* d_ws, size_t ws_size,
                              hipStream_t stream) {
    const float*         xs     = (const float*)d_in[0];
    const unsigned char* resets = (const unsigned char*)d_in[1]; // jnp bool_ = 1 byte
    const float*         Wx     = (const float*)d_in[2];
    const float*         Wh     = (const float*)d_in[3];
    const float*         b      = (const float*)d_in[4];
    float*               out    = (float*)d_out;

    // workspace layout (bytes, all 16B aligned), total ~103 MB
    const size_t off_wcat = 0;                                   // L*K*N bf16 = 33.5 MB
    const size_t off_bias = off_wcat + (size_t)LAYERS * KDIM * NDIM * 2;
    const size_t off_h    = off_bias + (size_t)LAYERS * NDIM * 4;        // 2*L*B*W bf16
    const size_t off_c    = off_h    + (size_t)2 * LAYERS * BW * 2;      // L*B*W fp32
    const size_t off_ymid = off_c    + (size_t)LAYERS * BW * 4;          // B*W bf16
    const size_t off_xsbf = off_ymid + (size_t)BW * 2;                   // T*B*W bf16
    char* ws = (char*)d_ws;
    unsigned short* wcat = (unsigned short*)(ws + off_wcat);
    float*          bias = (float*)(ws + off_bias);
    unsigned short* hbuf = (unsigned short*)(ws + off_h);
    float*          cbuf = (float*)(ws + off_c);
    unsigned short* ymid = (unsigned short*)(ws + off_ymid);
    unsigned short* xsbf = (unsigned short*)(ws + off_xsbf);

    convert_weights_kernel<<<4096, 256, 0, stream>>>(Wx, Wh, wcat);
    convert_xs_kernel<<<8192, 256, 0, stream>>>(xs, (unsigned*)xsbf);
    init_states_kernel<<<(2 * LAYERS * BW + 255) / 256, 256, 0, stream>>>(
        b, bias, hbuf, cbuf);

    const size_t wl = (size_t)KDIM * NDIM;   // wcat layer stride (elements)
    for (int t = 0; t < T_STEPS; ++t) {
        const unsigned char* r_t = resets + (size_t)t * BATCH;
        const int pin = t & 1, pout = (t + 1) & 1;

        // layer 0: x from pre-converted xs (bf16), y -> ymid (bf16)
        lstm_step_kernel<false><<<NDIM / 64, 256, 0, stream>>>(
            xsbf + (size_t)t * BW, r_t,
            wcat + 0 * wl, bias + 0 * NDIM,
            hbuf + ((size_t)pin  * LAYERS + 0) * BW,
            hbuf + ((size_t)pout * LAYERS + 0) * BW,
            cbuf + (size_t)0 * BW,
            ymid, nullptr);

        // layer 1: x from ymid (bf16), y -> out[t] (fp32)
        lstm_step_kernel<true><<<NDIM / 64, 256, 0, stream>>>(
            ymid, r_t,
            wcat + 1 * wl, bias + 1 * NDIM,
            hbuf + ((size_t)pin  * LAYERS + 1) * BW,
            hbuf + ((size_t)pout * LAYERS + 1) * BW,
            cbuf + (size_t)1 * BW,
            nullptr, out + (size_t)t * BW);
    }
}
